// GCN_3633542333139
// MI455X (gfx1250) — matrix operand order repaired and checked by reading, not compiled, not run on hardware
//
#include <hip/hip_runtime.h>

typedef __attribute__((ext_vector_type(16))) __bf16 v16bf;
typedef __attribute__((ext_vector_type(8)))  float  v8f;

#define F_IN 1024
#define KSTEPS (F_IN / 32)   // 32 K-steps of 32

// ---------------------------------------------------------------------------
// Layer-1 GEMM: hlin[N,16] = x[N,1024] @ W1[1024,16], bf16 WMMA, f32 accum.
// One wave computes a 16-row x 16-col output tile; 8 waves per block.
// W1 is converted to bf16 AND pre-swizzled into the per-lane B-fragment
// layout in LDS, so the inner loop issues one contiguous 32B LDS load per
// K-step instead of 16 strided 16-bit loads.
// ---------------------------------------------------------------------------
__global__ __launch_bounds__(256) void gemm1_wmma(
    const float* __restrict__ x, const float* __restrict__ W1,
    float* __restrict__ hlin) {
  // w1p[step][lane][j]: j-th bf16 of lane's B fragment for K-step `step`.
  __shared__ __align__(32) __bf16 w1p[KSTEPS * 32 * 16];  // 32 KB

  // Cooperative pack: B fragment element j of lane (half=lane>>4,
  // col=lane&15) is W1[(step*32 + (j&7) + 8*half + 16*(j>>3)) * 16 + col].
  for (int i = threadIdx.x; i < KSTEPS * 32 * 16; i += blockDim.x) {
    int step = i >> 9;          // / 512
    int rem  = i & 511;
    int ln   = rem >> 4;
    int j    = rem & 15;
    int hf   = ln >> 4;
    int cl   = ln & 15;
    int k    = step * 32 + (j & 7) + 8 * hf + 16 * (j >> 3);
    w1p[i] = (__bf16)W1[k * 16 + cl];
  }
  __syncthreads();

  const int lane = threadIdx.x & 31;
  const int wave = threadIdx.x >> 5;
  const int half = lane >> 4;       // which 16-lane half of the wave
  const int col  = lane & 15;       // A row / B,C,D column for this lane
  const int rowbase = (blockIdx.x * 8 + wave) * 16;
  const float* __restrict__ xrow = x + (size_t)(rowbase + col) * F_IN;
  const v16bf* __restrict__ bfrag = (const v16bf*)&w1p[lane * 16];

  v8f c = {};
#pragma unroll 2
  for (int k0 = 0; k0 < F_IN; k0 += 32) {
    // Prefetch ~4 steps ahead of this row (global_prefetch_b8).
    __builtin_prefetch(xrow + k0 + 128, 0, 1);

    // A fragment (16-bit A 16x32 layout): lanes 0-15 hold K = k0+{0..7,16..23},
    // lanes 16-31 hold K = k0+{8..15,24..31}. Two contiguous 32B chunks/lane.
    const float4* p0 = (const float4*)(xrow + k0 + 8 * half);
    const float4* p1 = (const float4*)(xrow + k0 + 16 + 8 * half);
    float4 q0 = p0[0], q1 = p0[1];
    float4 q2 = p1[0], q3 = p1[1];
    v16bf a;
    a[0]  = (__bf16)q0.x; a[1]  = (__bf16)q0.y; a[2]  = (__bf16)q0.z; a[3]  = (__bf16)q0.w;
    a[4]  = (__bf16)q1.x; a[5]  = (__bf16)q1.y; a[6]  = (__bf16)q1.z; a[7]  = (__bf16)q1.w;
    a[8]  = (__bf16)q2.x; a[9]  = (__bf16)q2.y; a[10] = (__bf16)q2.z; a[11] = (__bf16)q2.w;
    a[12] = (__bf16)q3.x; a[13] = (__bf16)q3.y; a[14] = (__bf16)q3.z; a[15] = (__bf16)q3.w;

    // B fragment: one contiguous 32B LDS load (pre-swizzled above).
    v16bf b = bfrag[(k0 >> 5) * 32];

#if defined(__gfx1250__)
    c = __builtin_amdgcn_wmma_f32_16x16x32_bf16(
        /*neg_a=*/false, a, /*neg_b=*/false, b,
        /*c_mod=*/(short)0, c, /*reuse_a=*/false, /*reuse_b=*/false);
#else
    (void)a; (void)b;
#endif
  }

  // D layout: VGPR r -> row M = r + 8*half, col N = lane&15.
#pragma unroll
  for (int r = 0; r < 8; ++r)
    hlin[(size_t)(rowbase + r + 8 * half) * 16 + col] = c[r];
}

// ---------------------------------------------------------------------------
// Helpers: zero-fill, degree accumulation, normalization factors.
// ---------------------------------------------------------------------------
__global__ void fill_zero(float* __restrict__ p, int n) {
  int i = blockIdx.x * blockDim.x + threadIdx.x;
  if (i < n) p[i] = 0.0f;
}

__global__ void deg_accum(const int* __restrict__ dst,
                          const float* __restrict__ ew,
                          float* __restrict__ deg, int E) {
  int e = blockIdx.x * blockDim.x + threadIdx.x;
  if (e < E) atomicAdd(&deg[dst[e]], ew[e]);
}

__global__ void deg_finalize(const float* __restrict__ deg,
                             float* __restrict__ dinv,
                             float* __restrict__ dinv2, int n) {
  int i = blockIdx.x * blockDim.x + threadIdx.x;
  if (i >= n) return;
  float r = rsqrtf(deg[i] + 1.0f);   // deg >= 0, +1 self loop -> no inf
  dinv[i]  = r;
  dinv2[i] = r * r;
}

// ---------------------------------------------------------------------------
// Edge scatter-add: agg[dst] += hlin[src] * (dinv[src]*ew*dinv[dst]).
// Feature maps fit in L2 (192MB), so the random atomics stay on-chip.
// ---------------------------------------------------------------------------
template <int F>
__global__ void edge_agg(const int* __restrict__ src, const int* __restrict__ dst,
                         const float* __restrict__ ew, const float* __restrict__ dinv,
                         const float* __restrict__ hlin, float* __restrict__ agg,
                         int E) {
  int e = blockIdx.x * blockDim.x + threadIdx.x;
  if (e >= E) return;
  int s = src[e], d = dst[e];
  float norm = dinv[s] * ew[e] * dinv[d];
#pragma unroll
  for (int f = 0; f < F; ++f)
    atomicAdd(&agg[(size_t)d * F + f], hlin[(size_t)s * F + f] * norm);
}

// Self-loop message + bias + ReLU, fused.
template <int F>
__global__ void node_finalize(const float* __restrict__ agg,
                              const float* __restrict__ hlin,
                              const float* __restrict__ dinv2,
                              const float* __restrict__ bias,
                              float* __restrict__ out, int total) {
  int i = blockIdx.x * blockDim.x + threadIdx.x;
  if (i >= total) return;
  int node = i / F, f = i - node * F;
  float v = agg[i] + hlin[i] * dinv2[node] + bias[f];
  out[i] = fmaxf(v, 0.0f);
}

// ---------------------------------------------------------------------------
// Skinny layers: 16->4 and 4->1 per-node FMAs (too narrow for 16x16 tiles).
// ---------------------------------------------------------------------------
__global__ void gemm2(const float* __restrict__ h1, const float* __restrict__ W2,
                      float* __restrict__ out, int n) {
  int i = blockIdx.x * blockDim.x + threadIdx.x;
  if (i >= n) return;
  float acc0 = 0.f, acc1 = 0.f, acc2 = 0.f, acc3 = 0.f;
  const float4* hp = (const float4*)(h1 + (size_t)i * 16);
#pragma unroll
  for (int kq = 0; kq < 4; ++kq) {
    float4 h = hp[kq];
    const float* w = W2 + kq * 16;
    acc0 += h.x * w[0] + h.y * w[4] + h.z * w[8]  + h.w * w[12];
    acc1 += h.x * w[1] + h.y * w[5] + h.z * w[9]  + h.w * w[13];
    acc2 += h.x * w[2] + h.y * w[6] + h.z * w[10] + h.w * w[14];
    acc3 += h.x * w[3] + h.y * w[7] + h.z * w[11] + h.w * w[15];
  }
  float4 r = {acc0, acc1, acc2, acc3};
  ((float4*)out)[i] = r;
}

__global__ void gemm3(const float* __restrict__ h2, const float* __restrict__ W3,
                      float* __restrict__ out, int n) {
  int i = blockIdx.x * blockDim.x + threadIdx.x;
  if (i >= n) return;
  float4 h = ((const float4*)h2)[i];
  out[i] = h.x * W3[0] + h.y * W3[1] + h.z * W3[2] + h.w * W3[3];
}

// ---------------------------------------------------------------------------
// Output GEMV: out[b] = flat . W_out[b,:] + b_out[b]. One block per output.
// ---------------------------------------------------------------------------
__global__ __launch_bounds__(256) void out_gemv(
    const float* __restrict__ flat, const float* __restrict__ Wout,
    const float* __restrict__ bout, float* __restrict__ out, int n) {
  int b = blockIdx.x;
  const float* wrow = Wout + (size_t)b * n;
  float acc = 0.f;
  for (int i = threadIdx.x; i < n; i += 256) acc += flat[i] * wrow[i];
  __shared__ float red[256];
  red[threadIdx.x] = acc;
  __syncthreads();
  for (int s = 128; s > 0; s >>= 1) {
    if (threadIdx.x < s) red[threadIdx.x] += red[threadIdx.x + s];
    __syncthreads();
  }
  if (threadIdx.x == 0) out[b] = red[0] + bout[b];
}

// ---------------------------------------------------------------------------
extern "C" void kernel_launch(void* const* d_in, const int* in_sizes, int n_in,
                              void* d_out, int out_size, void* d_ws, size_t ws_size,
                              hipStream_t stream) {
  const float* x    = (const float*)d_in[0];
  const int*   A    = (const int*)  d_in[1];
  const float* ew   = (const float*)d_in[2];
  const float* W1   = (const float*)d_in[3];
  const float* b1   = (const float*)d_in[4];
  const float* W2   = (const float*)d_in[5];
  const float* b2   = (const float*)d_in[6];
  const float* W3   = (const float*)d_in[7];
  const float* b3   = (const float*)d_in[8];
  const float* Wout = (const float*)d_in[9];
  const float* bout = (const float*)d_in[10];
  float* out = (float*)d_out;

  const int n   = in_sizes[0] / F_IN;   // 65536 nodes
  const int E   = in_sizes[1] / 2;      // 2097152 edges
  const int Bsz = in_sizes[10];         // 64 outputs
  const int* src = A;
  const int* dst = A + E;

  // Workspace layout (floats); buffers reused across layers.
  float* ws    = (float*)d_ws;
  float* hlin  = ws;                        // n*16 (linear output, pre-agg)
  float* agg   = hlin  + (size_t)n * 16;    // n*16 (scatter accumulator)
  float* hbuf  = agg   + (size_t)n * 16;    // n*16 (post-ReLU features)
  float* deg   = hbuf  + (size_t)n * 16;    // n
  float* dinv  = deg   + n;                 // n
  float* dinv2 = dinv  + n;                 // n

  const int T = 256;
  // Degrees / normalization (shared by all three layers).
  fill_zero<<<(n + T - 1) / T, T, 0, stream>>>(deg, n);
  deg_accum<<<(E + T - 1) / T, T, 0, stream>>>(dst, ew, deg, E);
  deg_finalize<<<(n + T - 1) / T, T, 0, stream>>>(deg, dinv, dinv2, n);

  // Layer 1: WMMA GEMM -> edge scatter -> self-loop+bias+ReLU.
  gemm1_wmma<<<n / 128, 256, 0, stream>>>(x, W1, hlin);
  fill_zero<<<(n * 16 + T - 1) / T, T, 0, stream>>>(agg, n * 16);
  edge_agg<16><<<(E + T - 1) / T, T, 0, stream>>>(src, dst, ew, dinv, hlin, agg, E);
  node_finalize<16><<<(n * 16 + T - 1) / T, T, 0, stream>>>(agg, hlin, dinv2, b1, hbuf, n * 16);

  // Layer 2 (16 -> 4).
  gemm2<<<(n + T - 1) / T, T, 0, stream>>>(hbuf, W2, hlin, n);
  fill_zero<<<(n * 4 + T - 1) / T, T, 0, stream>>>(agg, n * 4);
  edge_agg<4><<<(E + T - 1) / T, T, 0, stream>>>(src, dst, ew, dinv, hlin, agg, E);
  node_finalize<4><<<(n * 4 + T - 1) / T, T, 0, stream>>>(agg, hlin, dinv2, b2, hbuf, n * 4);

  // Layer 3 (4 -> 1).
  gemm3<<<(n + T - 1) / T, T, 0, stream>>>(hbuf, W3, hlin, n);
  fill_zero<<<(n + T - 1) / T, T, 0, stream>>>(agg, n);
  edge_agg<1><<<(E + T - 1) / T, T, 0, stream>>>(src, dst, ew, dinv, hlin, agg, E);
  node_finalize<1><<<(n + T - 1) / T, T, 0, stream>>>(agg, hlin, dinv2, b3, hbuf, n);

  // Final dense head: [N] x [B,N]^T + b_out.
  out_gemv<<<Bsz, 256, 0, stream>>>(hbuf, Wout, bout, out, n);
}
